// BRGCN_89635967467723
// MI455X (gfx1250) — matrix operand order repaired and checked by reading, not compiled, not run on hardware
//
#include <hip/hip_runtime.h>
#include <hip/hip_bf16.h>
#include <math.h>

typedef __bf16 bf16_t;
typedef __attribute__((ext_vector_type(16))) __bf16 v16bf;
typedef __attribute__((ext_vector_type(8)))  float  v8f;
typedef __attribute__((ext_vector_type(4)))  float  v4f;

#define R_   5
#define N0_  100000
#define N1_  30000
#define N2_  15000
#define E0_  300000
#define E1_  150000
#define IN_  128
#define HC0_ 256
#define H0_  4
#define C0_  64
#define C1_  40
#define C1P_ 64

// ---------- helpers ----------
__device__ __forceinline__ float wred_sum(float v) {
#pragma unroll
  for (int m = 16; m >= 1; m >>= 1) v += __shfl_xor(v, m, 32);
  return v;
}
__device__ __forceinline__ float wred_max(float v) {
#pragma unroll
  for (int m = 16; m >= 1; m >>= 1) v = fmaxf(v, __shfl_xor(v, m, 32));
  return v;
}
// order-preserving float<->uint encoding for atomic max
__device__ __forceinline__ unsigned fenc(float f) {
  unsigned u = __float_as_uint(f);
  return (u & 0x80000000u) ? ~u : (u | 0x80000000u);
}
__device__ __forceinline__ float fdec(unsigned u) {
  unsigned b = (u & 0x80000000u) ? (u & 0x7FFFFFFFu) : ~u;
  return __uint_as_float(b);
}

// ---------- gather + cast ----------
__global__ void k_gather(const int* __restrict__ n_id, const int* __restrict__ lidx,
                         const float* __restrict__ emb, bf16_t* __restrict__ xbf) {
  int i = blockIdx.x;
  int c = threadIdx.x;              // 128 threads
  int g = lidx[n_id[i]];
  xbf[(size_t)i * IN_ + c] = (bf16_t)emb[(size_t)g * IN_ + c];
}

// ---------- weight transpose + pad to bf16 [Np,Kp] ----------
__global__ void k_wt(const float* __restrict__ W, bf16_t* __restrict__ WT,
                     int K, int N, int Kp, int Np) {
  int idx = blockIdx.x * blockDim.x + threadIdx.x;
  if (idx >= Np * Kp) return;
  int n = idx / Kp, k = idx % Kp;
  float v = (n < N && k < K) ? W[(size_t)k * N + n] : 0.0f;
  WT[idx] = (bf16_t)v;
}

// ---------- bf16 WMMA GEMM: C[M,Np] = A[M,KP] * BT[Np,KP]^T ----------
// One wave owns a 16x64 strip (4 accumulators). KP compile-time -> full unroll,
// distinct B fragment registers per tile -> loads pipeline across WMMAs.
template<int KP>
__global__ __launch_bounds__(256)
void k_gemm_bf16(const bf16_t* __restrict__ A, const bf16_t* __restrict__ BT,
                 float* __restrict__ Cf, bf16_t* __restrict__ Cb,
                 int M, int Np) {
  const int lane = threadIdx.x & 31;
  const int wid  = (blockIdx.x * blockDim.x + threadIdx.x) >> 5;
  const int nstrips = Np >> 6;                // 64 cols per wave
  const int tiles_m = (M + 15) >> 4;
  const int tm = wid / nstrips;
  const int tn = wid % nstrips;
  if (tm >= tiles_m) return;

  const int m0 = tm << 4;
  int rowa = m0 + (lane & 15);
  if (rowa > M - 1) rowa = M - 1;             // clamp (stores guarded)
  const int klo  = (lane < 16) ? 0 : 8;       // A frag K offset
  const int kgrp = (lane < 16) ? 0 : 8;       // B frag K offset
  const int n0 = tn << 6;
  const int coll = lane & 15;

  const bf16_t* aprow = A + (size_t)rowa * KP + klo;
  const bf16_t* bbase = BT + (size_t)(n0 + coll) * KP + kgrp;

  v8f acc0 = (v8f){0.f,0.f,0.f,0.f,0.f,0.f,0.f,0.f};
  v8f acc1 = acc0, acc2 = acc0, acc3 = acc0;

#pragma unroll
  for (int k0 = 0; k0 < KP; k0 += 32) {
    v16bf a, b0, b1, b2, b3;
    ((v4f*)&a)[0]  = *(const v4f*)(aprow + k0);            // K = klo..klo+7
    ((v4f*)&a)[1]  = *(const v4f*)(aprow + k0 + 16);       // K = 16+klo..
    const bf16_t* bp0 = bbase + k0;
    const bf16_t* bp1 = bp0 + 16 * KP;
    const bf16_t* bp2 = bp0 + 32 * KP;
    const bf16_t* bp3 = bp0 + 48 * KP;
    ((v4f*)&b0)[0] = *(const v4f*)(bp0);
    ((v4f*)&b0)[1] = *(const v4f*)(bp0 + 16);
    ((v4f*)&b1)[0] = *(const v4f*)(bp1);
    ((v4f*)&b1)[1] = *(const v4f*)(bp1 + 16);
    ((v4f*)&b2)[0] = *(const v4f*)(bp2);
    ((v4f*)&b2)[1] = *(const v4f*)(bp2 + 16);
    ((v4f*)&b3)[0] = *(const v4f*)(bp3);
    ((v4f*)&b3)[1] = *(const v4f*)(bp3 + 16);
    acc0 = __builtin_amdgcn_wmma_f32_16x16x32_bf16(false, a, false, b0, (short)0, acc0, false, false);
    acc1 = __builtin_amdgcn_wmma_f32_16x16x32_bf16(false, a, false, b1, (short)0, acc1, false, false);
    acc2 = __builtin_amdgcn_wmma_f32_16x16x32_bf16(false, a, false, b2, (short)0, acc2, false, false);
    acc3 = __builtin_amdgcn_wmma_f32_16x16x32_bf16(false, a, false, b3, (short)0, acc3, false, false);
  }

  const int rbase = m0 + ((lane < 16) ? 0 : 8);
  v8f accs[4] = {acc0, acc1, acc2, acc3};
#pragma unroll
  for (int t = 0; t < 4; ++t) {
    const int cc = n0 + (t << 4) + coll;
#pragma unroll
    for (int j = 0; j < 8; ++j) {
      int r = rbase + j;
      if (r < M) {
        float v = accs[t][j];
        if (Cb) Cb[(size_t)r * Np + cc] = (bf16_t)v;
        else    Cf[(size_t)r * Np + cc] = v;
      }
    }
  }
}

// ---------- fill uint ----------
__global__ void k_fill_u32(unsigned* __restrict__ p, int n, unsigned v) {
  int i = blockIdx.x * blockDim.x + threadIdx.x;
  if (i < n) p[i] = v;
}

// ---------- per-edge attention logits + segment max ----------
__global__ void k_edge_logits(const int* __restrict__ src, const int* __restrict__ tgt,
                              const int* __restrict__ et,
                              const float* __restrict__ hj, const float* __restrict__ hi,
                              const float* __restrict__ attj, const float* __restrict__ atti,
                              float* __restrict__ alpha, unsigned* __restrict__ segmax,
                              int E, int H, int C, int Sj, int Si) {
  int idx = blockIdx.x * blockDim.x + threadIdx.x;
  if (idx >= E * H) return;
  int e = idx / H, h = idx % H;
  int s = src[e], t = tgt[e], r = et[e];
  const float* aj = attj + ((size_t)r * H + h) * C;
  const float* ai = atti + ((size_t)r * H + h) * C;
  const float* xj = hj + (size_t)s * Sj + h * C;
  const float* xi = hi + (size_t)t * Si + h * C;
  float acc = 0.f;
  for (int c = 0; c < C; ++c) acc += aj[c] * xj[c] + ai[c] * xi[c];
  float a = (acc > 0.f) ? acc : 0.2f * acc;   // leaky relu
  alpha[idx] = a;
  int seg = t * R_ + r;
  atomicMax(&segmax[(size_t)seg * H + h], fenc(a));
}

// ---------- exp + segment sum ----------
__global__ void k_edge_exp(const int* __restrict__ tgt, const int* __restrict__ et,
                           float* __restrict__ alpha, const unsigned* __restrict__ segmax,
                           float* __restrict__ den, int E, int H) {
  int idx = blockIdx.x * blockDim.x + threadIdx.x;
  if (idx >= E * H) return;
  int e = idx / H, h = idx % H;
  int seg = tgt[e] * R_ + et[e];
  float m = fdec(segmax[(size_t)seg * H + h]);
  float ex = __expf(alpha[idx] - m);
  alpha[idx] = ex;
  atomicAdd(&den[(size_t)seg * H + h], ex);
}

// ---------- weighted scatter into z ----------
__global__ void k_edge_scatter(const int* __restrict__ src, const int* __restrict__ tgt,
                               const int* __restrict__ et,
                               const float* __restrict__ ex, const float* __restrict__ den,
                               const float* __restrict__ hj, float* __restrict__ z,
                               int E, int HCv, int H, int C, int Sj, int Sz, int N) {
  long long idx = (long long)blockIdx.x * blockDim.x + threadIdx.x;
  if (idx >= (long long)E * HCv) return;
  int e = (int)(idx / HCv), c = (int)(idx % HCv);
  int h = c / C;
  int s = src[e], t = tgt[e], r = et[e];
  int seg = t * R_ + r;
  float w = ex[(size_t)e * H + h] / fmaxf(den[(size_t)seg * H + h], 1e-16f);
  atomicAdd(&z[((size_t)r * N + t) * Sz + c], w * hj[(size_t)s * Sj + c]);
}

// ---------- f32 -> bf16 convert ----------
__global__ void k_cvt(const float* __restrict__ in, bf16_t* __restrict__ out, long long n) {
  long long i = (long long)blockIdx.x * blockDim.x + threadIdx.x;
  if (i < n) out[i] = (bf16_t)in[i];
}

// ---------- layer0 bilevel attention + combine + relu -> x1_bf ----------
__global__ __launch_bounds__(256)
void k_attn0(const bf16_t* __restrict__ q, const bf16_t* __restrict__ k,
             const bf16_t* __restrict__ v, const float* __restrict__ selfx,
             const float* __restrict__ sb, const float* __restrict__ Wrel,
             bf16_t* __restrict__ x1bf) {
  int wid = (blockIdx.x * blockDim.x + threadIdx.x) >> 5;
  int lane = threadIdx.x & 31;
  if (wid >= N1_ * H0_) return;
  int n = wid >> 2, h = wid & 3;

  float qr[R_][2], kr[R_][2], vr[R_][2];
#pragma unroll
  for (int r = 0; r < R_; ++r) {
    size_t base = ((size_t)r * N1_ + n) * HC0_ + h * C0_ + lane;
    qr[r][0] = (float)q[base];      qr[r][1] = (float)q[base + 32];
    kr[r][0] = (float)k[base];      kr[r][1] = (float)k[base + 32];
    vr[r][0] = (float)v[base];      vr[r][1] = (float)v[base + 32];
  }
  float psi[R_][R_];
#pragma unroll
  for (int r = 0; r < R_; ++r)
#pragma unroll
    for (int s = 0; s < R_; ++s)
      psi[r][s] = wred_sum(qr[r][0] * kr[s][0] + qr[r][1] * kr[s][1]);

  float g[R_] = {0.f, 0.f, 0.f, 0.f, 0.f};
#pragma unroll
  for (int r = 0; r < R_; ++r) {
    float rowsum = 0.f;
#pragma unroll
    for (int s = 0; s < R_; ++s) rowsum += psi[r][s];
    float vals[R_]; float mx = -INFINITY;
#pragma unroll
    for (int s = 0; s < R_; ++s) {
      float p = psi[r][s];
      if (p == 0.f && rowsum != 0.f) p = -INFINITY;
      vals[s] = p; mx = fmaxf(mx, p);
    }
    float se = 0.f;
#pragma unroll
    for (int s = 0; s < R_; ++s) { vals[s] = __expf(vals[s] - mx); se += vals[s]; }
    float wr = Wrel[r] / se;
#pragma unroll
    for (int s = 0; s < R_; ++s) g[s] += wr * vals[s];
  }
#pragma unroll
  for (int j = 0; j < 2; ++j) {
    float o = 0.f;
#pragma unroll
    for (int s = 0; s < R_; ++s) o += g[s] * vr[s][j];
    int c = h * C0_ + lane + j * 32;
    o += selfx[(size_t)n * HC0_ + c] + sb[c];
    o = fmaxf(o, 0.f);                          // relu
    x1bf[(size_t)n * HC0_ + c] = (bf16_t)o;
  }
}

// ---------- layer1 bilevel attention + combine + log_softmax -> out ----------
__global__ __launch_bounds__(256)
void k_attn1(const bf16_t* __restrict__ q, const bf16_t* __restrict__ k,
             const bf16_t* __restrict__ v, const float* __restrict__ selfx,
             const float* __restrict__ sb, const float* __restrict__ Wrel,
             float* __restrict__ out) {
  int n = (blockIdx.x * blockDim.x + threadIdx.x) >> 5;
  int lane = threadIdx.x & 31;
  if (n >= N2_) return;
  int c0 = lane, c1 = lane + 32;
  bool ok1 = (c1 < C1_);

  float qr[R_][2], kr[R_][2], vr[R_][2];
#pragma unroll
  for (int r = 0; r < R_; ++r) {
    size_t base = ((size_t)r * N2_ + n) * C1P_;
    qr[r][0] = (float)q[base + c0]; qr[r][1] = (float)q[base + c1]; // pad cols are 0
    kr[r][0] = (float)k[base + c0]; kr[r][1] = (float)k[base + c1];
    vr[r][0] = (float)v[base + c0]; vr[r][1] = (float)v[base + c1];
  }
  float psi[R_][R_];
#pragma unroll
  for (int r = 0; r < R_; ++r)
#pragma unroll
    for (int s = 0; s < R_; ++s)
      psi[r][s] = wred_sum(qr[r][0] * kr[s][0] + qr[r][1] * kr[s][1]);

  float g[R_] = {0.f, 0.f, 0.f, 0.f, 0.f};
#pragma unroll
  for (int r = 0; r < R_; ++r) {
    float rowsum = 0.f;
#pragma unroll
    for (int s = 0; s < R_; ++s) rowsum += psi[r][s];
    float vals[R_]; float mx = -INFINITY;
#pragma unroll
    for (int s = 0; s < R_; ++s) {
      float p = psi[r][s];
      if (p == 0.f && rowsum != 0.f) p = -INFINITY;
      vals[s] = p; mx = fmaxf(mx, p);
    }
    float se = 0.f;
#pragma unroll
    for (int s = 0; s < R_; ++s) { vals[s] = __expf(vals[s] - mx); se += vals[s]; }
    float wr = Wrel[r] / se;
#pragma unroll
    for (int s = 0; s < R_; ++s) g[s] += wr * vals[s];
  }
  float o0 = 0.f, o1 = 0.f;
#pragma unroll
  for (int s = 0; s < R_; ++s) { o0 += g[s] * vr[s][0]; o1 += g[s] * vr[s][1]; }
  o0 += selfx[(size_t)n * C1P_ + c0] + sb[c0];
  if (ok1) o1 += selfx[(size_t)n * C1P_ + c1] + sb[c1];

  float mx = wred_max(fmaxf(o0, ok1 ? o1 : -INFINITY));
  float es = wred_sum(__expf(o0 - mx) + (ok1 ? __expf(o1 - mx) : 0.f));
  float l = __logf(es);
  out[(size_t)n * C1_ + c0] = o0 - mx - l;
  if (ok1) out[(size_t)n * C1_ + c1] = o1 - mx - l;
}

// ======================================================================
extern "C" void kernel_launch(void* const* d_in, const int* in_sizes, int n_in,
                              void* d_out, int out_size, void* d_ws, size_t ws_size,
                              hipStream_t stream) {
  (void)in_sizes; (void)n_in; (void)out_size; (void)ws_size;
  const int*   n_id  = (const int*)d_in[0];
  const int*   lidx  = (const int*)d_in[1];
  const int*   ei0   = (const int*)d_in[3];
  const int*   et0   = (const int*)d_in[4];
  const int*   ei1   = (const int*)d_in[5];
  const int*   et1   = (const int*)d_in[6];
  const float* emb   = (const float*)d_in[7];
  const float* Wj0   = (const float*)d_in[8];
  const float* Wi0   = (const float*)d_in[9];
  const float* attj0 = (const float*)d_in[10];
  const float* atti0 = (const float*)d_in[11];
  const float* Wq0   = (const float*)d_in[12];
  const float* Wk0   = (const float*)d_in[13];
  const float* Wv0   = (const float*)d_in[14];
  const float* sw0   = (const float*)d_in[15];
  const float* sb0   = (const float*)d_in[16];
  const float* Wrel0 = (const float*)d_in[17];
  const float* Wj1   = (const float*)d_in[18];
  const float* Wi1   = (const float*)d_in[19];
  const float* attj1 = (const float*)d_in[20];
  const float* atti1 = (const float*)d_in[21];
  const float* Wq1   = (const float*)d_in[22];
  const float* Wk1   = (const float*)d_in[23];
  const float* Wv1   = (const float*)d_in[24];
  const float* sw1   = (const float*)d_in[25];
  const float* sb1   = (const float*)d_in[26];
  const float* Wrel1 = (const float*)d_in[27];
  float* out = (float*)d_out;

  char* ws = (char*)d_ws;
  size_t off = 0;
  auto alloc = [&](size_t bytes) -> char* {
    char* p = ws + off;
    off = (off + bytes + 255) & ~(size_t)255;
    return p;
  };

  bf16_t*   xbf    = (bf16_t*)  alloc((size_t)N0_ * IN_ * 2);
  float*    hj0    = (float*)   alloc((size_t)N0_ * HC0_ * 4);
  float*    hi0    = (float*)   alloc((size_t)N1_ * HC0_ * 4);
  float*    self0  = (float*)   alloc((size_t)N1_ * HC0_ * 4);
  float*    alpha0 = (float*)   alloc((size_t)E0_ * H0_ * 4);
  unsigned* smax0  = (unsigned*)alloc((size_t)N1_ * R_ * H0_ * 4);
  float*    den0   = (float*)   alloc((size_t)N1_ * R_ * H0_ * 4);
  float*    z0     = (float*)   alloc((size_t)R_ * N1_ * HC0_ * 4);
  bf16_t*   z0bf   = (bf16_t*)  alloc((size_t)R_ * N1_ * HC0_ * 2);
  bf16_t*   q0bf   = (bf16_t*)  alloc((size_t)R_ * N1_ * HC0_ * 2);
  bf16_t*   k0bf   = (bf16_t*)  alloc((size_t)R_ * N1_ * HC0_ * 2);
  bf16_t*   v0bf   = (bf16_t*)  alloc((size_t)R_ * N1_ * HC0_ * 2);
  bf16_t*   x1bf   = (bf16_t*)  alloc((size_t)N1_ * HC0_ * 2);
  float*    hj1    = (float*)   alloc((size_t)N1_ * C1P_ * 4);
  float*    hi1    = (float*)   alloc((size_t)N2_ * C1P_ * 4);
  float*    self1  = (float*)   alloc((size_t)N2_ * C1P_ * 4);
  float*    alpha1 = (float*)   alloc((size_t)E1_ * 4);
  unsigned* smax1  = (unsigned*)alloc((size_t)N2_ * R_ * 4);
  float*    den1   = (float*)   alloc((size_t)N2_ * R_ * 4);
  float*    z1     = (float*)   alloc((size_t)R_ * N2_ * C1P_ * 4);
  bf16_t*   z1bf   = (bf16_t*)  alloc((size_t)R_ * N2_ * C1P_ * 2);
  bf16_t*   q1bf   = (bf16_t*)  alloc((size_t)R_ * N2_ * C1P_ * 2);
  bf16_t*   k1bf   = (bf16_t*)  alloc((size_t)R_ * N2_ * C1P_ * 2);
  bf16_t*   v1bf   = (bf16_t*)  alloc((size_t)R_ * N2_ * C1P_ * 2);
  bf16_t*   Wj0T   = (bf16_t*)  alloc((size_t)HC0_ * IN_ * 2);
  bf16_t*   Wi0T   = (bf16_t*)  alloc((size_t)HC0_ * IN_ * 2);
  bf16_t*   sw0T   = (bf16_t*)  alloc((size_t)HC0_ * IN_ * 2);
  bf16_t*   Wq0T   = (bf16_t*)  alloc((size_t)R_ * HC0_ * HC0_ * 2);
  bf16_t*   Wk0T   = (bf16_t*)  alloc((size_t)R_ * HC0_ * HC0_ * 2);
  bf16_t*   Wv0T   = (bf16_t*)  alloc((size_t)R_ * HC0_ * HC0_ * 2);
  bf16_t*   Wj1T   = (bf16_t*)  alloc((size_t)C1P_ * HC0_ * 2);
  bf16_t*   Wi1T   = (bf16_t*)  alloc((size_t)C1P_ * HC0_ * 2);
  bf16_t*   sw1T   = (bf16_t*)  alloc((size_t)C1P_ * HC0_ * 2);
  bf16_t*   Wq1T   = (bf16_t*)  alloc((size_t)R_ * C1P_ * C1P_ * 2);
  bf16_t*   Wk1T   = (bf16_t*)  alloc((size_t)R_ * C1P_ * C1P_ * 2);
  bf16_t*   Wv1T   = (bf16_t*)  alloc((size_t)R_ * C1P_ * C1P_ * 2);

  // zero accumulation buffers (every call: graph replays reuse them)
  hipMemsetAsync(z0,   0, (size_t)R_ * N1_ * HC0_ * 4, stream);
  hipMemsetAsync(z1,   0, (size_t)R_ * N2_ * C1P_ * 4, stream);
  hipMemsetAsync(den0, 0, (size_t)N1_ * R_ * H0_ * 4, stream);
  hipMemsetAsync(den1, 0, (size_t)N2_ * R_ * 4, stream);
  k_fill_u32<<<(N1_ * R_ * H0_ + 255) / 256, 256, 0, stream>>>(smax0, N1_ * R_ * H0_, 0x007FFFFFu);
  k_fill_u32<<<(N2_ * R_ + 255) / 256, 256, 0, stream>>>(smax1, N2_ * R_, 0x007FFFFFu);

  auto wt = [&](const float* W, bf16_t* WT, int K, int N, int Kp, int Np) {
    int cnt = Np * Kp;
    k_wt<<<(cnt + 255) / 256, 256, 0, stream>>>(W, WT, K, N, Kp, Np);
  };
  auto gemm = [&](const bf16_t* A, const bf16_t* BT, float* Cf, bf16_t* Cb,
                  int M, int Kp, int Np) {
    int tiles = ((M + 15) / 16) * (Np / 64);
    int blocks = (tiles + 7) / 8;
    if (Kp == 128)      k_gemm_bf16<128><<<blocks, 256, 0, stream>>>(A, BT, Cf, Cb, M, Np);
    else if (Kp == 256) k_gemm_bf16<256><<<blocks, 256, 0, stream>>>(A, BT, Cf, Cb, M, Np);
    else                k_gemm_bf16<64><<<blocks, 256, 0, stream>>>(A, BT, Cf, Cb, M, Np);
  };

  // ---- weight prep ----
  wt(Wj0, Wj0T, IN_, HC0_, IN_, HC0_);
  wt(Wi0, Wi0T, IN_, HC0_, IN_, HC0_);
  wt(sw0, sw0T, IN_, HC0_, IN_, HC0_);
  for (int r = 0; r < R_; ++r) {
    wt(Wq0 + (size_t)r * HC0_ * HC0_, Wq0T + (size_t)r * HC0_ * HC0_, HC0_, HC0_, HC0_, HC0_);
    wt(Wk0 + (size_t)r * HC0_ * HC0_, Wk0T + (size_t)r * HC0_ * HC0_, HC0_, HC0_, HC0_, HC0_);
    wt(Wv0 + (size_t)r * HC0_ * HC0_, Wv0T + (size_t)r * HC0_ * HC0_, HC0_, HC0_, HC0_, HC0_);
  }
  wt(Wj1, Wj1T, HC0_, C1_, HC0_, C1P_);
  wt(Wi1, Wi1T, HC0_, C1_, HC0_, C1P_);
  wt(sw1, sw1T, HC0_, C1_, HC0_, C1P_);
  for (int r = 0; r < R_; ++r) {
    wt(Wq1 + (size_t)r * C1_ * C1_, Wq1T + (size_t)r * C1P_ * C1P_, C1_, C1_, C1P_, C1P_);
    wt(Wk1 + (size_t)r * C1_ * C1_, Wk1T + (size_t)r * C1P_ * C1P_, C1_, C1_, C1P_, C1P_);
    wt(Wv1 + (size_t)r * C1_ * C1_, Wv1T + (size_t)r * C1P_ * C1P_, C1_, C1_, C1P_, C1P_);
  }

  // ---- gather ----
  k_gather<<<N0_, IN_, 0, stream>>>(n_id, lidx, emb, xbf);

  // ================= layer 0 =================
  gemm(xbf, Wj0T, hj0, nullptr, N0_, IN_, HC0_);
  gemm(xbf, Wi0T, hi0, nullptr, N1_, IN_, HC0_);
  gemm(xbf, sw0T, self0, nullptr, N1_, IN_, HC0_);

  const int* src0 = ei0; const int* tgt0 = ei0 + E0_;
  int eh0 = E0_ * H0_;
  k_edge_logits<<<(eh0 + 255) / 256, 256, 0, stream>>>(src0, tgt0, et0, hj0, hi0,
      attj0, atti0, alpha0, smax0, E0_, H0_, C0_, HC0_, HC0_);
  k_edge_exp<<<(eh0 + 255) / 256, 256, 0, stream>>>(tgt0, et0, alpha0, smax0, den0, E0_, H0_);
  long long sc0 = (long long)E0_ * HC0_;
  k_edge_scatter<<<(unsigned)((sc0 + 255) / 256), 256, 0, stream>>>(src0, tgt0, et0,
      alpha0, den0, hj0, z0, E0_, HC0_, H0_, C0_, HC0_, HC0_, N1_);

  long long zc0 = (long long)R_ * N1_ * HC0_;
  k_cvt<<<(unsigned)((zc0 + 255) / 256), 256, 0, stream>>>(z0, z0bf, zc0);

  for (int r = 0; r < R_; ++r) {
    const bf16_t* zr = z0bf + (size_t)r * N1_ * HC0_;
    gemm(zr, Wq0T + (size_t)r * HC0_ * HC0_, nullptr, q0bf + (size_t)r * N1_ * HC0_, N1_, HC0_, HC0_);
    gemm(zr, Wk0T + (size_t)r * HC0_ * HC0_, nullptr, k0bf + (size_t)r * N1_ * HC0_, N1_, HC0_, HC0_);
    gemm(zr, Wv0T + (size_t)r * HC0_ * HC0_, nullptr, v0bf + (size_t)r * N1_ * HC0_, N1_, HC0_, HC0_);
  }

  int waves0 = N1_ * H0_;
  k_attn0<<<(waves0 * 32 + 255) / 256, 256, 0, stream>>>(q0bf, k0bf, v0bf, self0, sb0, Wrel0, x1bf);

  // ================= layer 1 =================
  gemm(x1bf, Wj1T, hj1, nullptr, N1_, HC0_, C1P_);
  gemm(x1bf, Wi1T, hi1, nullptr, N2_, HC0_, C1P_);
  gemm(x1bf, sw1T, self1, nullptr, N2_, HC0_, C1P_);

  const int* src1 = ei1; const int* tgt1 = ei1 + E1_;
  k_edge_logits<<<(E1_ + 255) / 256, 256, 0, stream>>>(src1, tgt1, et1, hj1, hi1,
      attj1, atti1, alpha1, smax1, E1_, 1, C1_, C1P_, C1P_);
  k_edge_exp<<<(E1_ + 255) / 256, 256, 0, stream>>>(tgt1, et1, alpha1, smax1, den1, E1_, 1);
  long long sc1 = (long long)E1_ * C1_;
  k_edge_scatter<<<(unsigned)((sc1 + 255) / 256), 256, 0, stream>>>(src1, tgt1, et1,
      alpha1, den1, hj1, z1, E1_, C1_, 1, C1_, C1P_, C1P_, N2_);

  long long zc1 = (long long)R_ * N2_ * C1P_;
  k_cvt<<<(unsigned)((zc1 + 255) / 256), 256, 0, stream>>>(z1, z1bf, zc1);

  for (int r = 0; r < R_; ++r) {
    const bf16_t* zr = z1bf + (size_t)r * N2_ * C1P_;
    gemm(zr, Wq1T + (size_t)r * C1P_ * C1P_, nullptr, q1bf + (size_t)r * N2_ * C1P_, N2_, C1P_, C1P_);
    gemm(zr, Wk1T + (size_t)r * C1P_ * C1P_, nullptr, k1bf + (size_t)r * N2_ * C1P_, N2_, C1P_, C1P_);
    gemm(zr, Wv1T + (size_t)r * C1P_ * C1P_, nullptr, v1bf + (size_t)r * N2_ * C1P_, N2_, C1P_, C1P_);
  }

  k_attn1<<<(N2_ * 32 + 255) / 256, 256, 0, stream>>>(q1bf, k1bf, v1bf, self1, sb1, Wrel1, out);
}
